// UniversalBehavioralTransformer_86887188398570
// MI455X (gfx1250) — matrix operand order repaired and checked by reading, not compiled, not run on hardware
//
#include <hip/hip_runtime.h>
#include <hip/hip_bf16.h>

typedef __attribute__((ext_vector_type(16))) __bf16        v16bf;
typedef __attribute__((ext_vector_type(8)))  float         v8f;
typedef __attribute__((ext_vector_type(8)))  unsigned int  v8u;

constexpr int CB = 16, CS = 512, CH = 512, CNH = 8, CHD = 64, CFF = 2048, CNT = 5;
constexpr int CTOK = CB * CS;

__device__ __forceinline__ unsigned short f2bf(float f) {
  unsigned int u = __float_as_uint(f);
  u += 0x7FFFu + ((u >> 16) & 1u);
  return (unsigned short)(u >> 16);
}
__device__ __forceinline__ float bf2f(unsigned short h) {
  return __uint_as_float(((unsigned int)h) << 16);
}

// ---------------------------------------------------------------------------
// LDS-staged batched bf16 WMMA GEMM: C[M,N] = alpha * A[M,K] @ Bt[N,K]^T (+bias)
// Block = 8 waves -> 128x64 output tile. B tile (64x32k = 4KB) is staged into
// LDS with GLOBAL_LOAD_ASYNC_TO_LDS_B128 (double-buffered, ASYNCcnt-tracked)
// and shared by all 8 waves; A fragments stream from global per wave.
// Requires M % 128 == 0, N % 64 == 0, K % 32 == 0.
// ---------------------------------------------------------------------------
__global__ __launch_bounds__(256) void gemm_bf16_lds(
    const unsigned short* __restrict__ A, long long lda,
    const unsigned short* __restrict__ Bt, long long ldb,
    const float* __restrict__ bias,
    float* __restrict__ Cf, unsigned short* __restrict__ Cbf, long long ldc,
    int M, int N, int K, float alpha, int relu, int zdiv,
    long long sA1, long long sA2, long long sB1, long long sB2,
    long long sC1, long long sC2)
{
  __shared__ unsigned short tileB[2][64 * 32] __attribute__((aligned(32)));

  int z  = blockIdx.z;
  int zq = z / zdiv, zr = z % zdiv;
  A  += (long long)zq * sA1 + (long long)zr * sA2;
  Bt += (long long)zq * sB1 + (long long)zr * sB2;
  long long cOff = (long long)zq * sC1 + (long long)zr * sC2;

  int mblocks = M >> 7;
  int bx = blockIdx.x;
  int tmb = bx % mblocks, tn = bx / mblocks;
  int wid = threadIdx.x >> 5, lane = threadIdx.x & 31;
  int row0 = (tmb << 7) + (wid << 4);
  int col0 = tn << 6;
  int mrow = lane & 15, half = lane >> 4;

  // async staging assignment: one b128 per thread per k-step
  int sn = threadIdx.x >> 2;           // 0..63 : row of Bt strip (N dim)
  int sk = (threadIdx.x & 3) << 3;     // 0,8,16,24 : k offset (ushorts)
  const unsigned short* gB = Bt + (long long)(col0 + sn) * ldb + sk;
  unsigned lds0 = (unsigned)(unsigned long long)(uintptr_t)&tileB[0][sn * 32 + sk];
  unsigned lds1 = (unsigned)(unsigned long long)(uintptr_t)&tileB[1][sn * 32 + sk];

  v8f a0 = {}, a1 = {}, a2 = {}, a3 = {};
  const unsigned short* aBase = A + (long long)(row0 + mrow) * lda + half * 8;

  // stage k-step 0 into buffer 0
  {
    unsigned long long ga = (unsigned long long)(uintptr_t)gB;
    asm volatile("global_load_async_to_lds_b128 %0, %1, off"
                 :: "v"(lds0), "v"(ga) : "memory");
  }

  int nsteps = K >> 5;
  for (int s = 0; s < nsteps; ++s) {
    asm volatile("s_wait_asynccnt 0x0" ::: "memory");
    __syncthreads();
    if (s + 1 < nsteps) {
      unsigned long long ga = (unsigned long long)(uintptr_t)(gB + ((s + 1) << 5));
      unsigned lo = ((s + 1) & 1) ? lds1 : lds0;
      asm volatile("global_load_async_to_lds_b128 %0, %1, off"
                   :: "v"(lo), "v"(ga) : "memory");
    }
    int k0 = s << 5;
    const unsigned short* ap = aBase + k0;
    v8u au;
    au[0] = *(const unsigned int*)(ap + 0);
    au[1] = *(const unsigned int*)(ap + 2);
    au[2] = *(const unsigned int*)(ap + 4);
    au[3] = *(const unsigned int*)(ap + 6);
    au[4] = *(const unsigned int*)(ap + 16);
    au[5] = *(const unsigned int*)(ap + 18);
    au[6] = *(const unsigned int*)(ap + 20);
    au[7] = *(const unsigned int*)(ap + 22);
    v16bf av = __builtin_bit_cast(v16bf, au);

    const unsigned short* tb = &tileB[s & 1][0];
    auto loadB = [&](int nt) -> v16bf {
      typedef __attribute__((ext_vector_type(4))) unsigned int v4u;
      const unsigned short* bp = tb + (((nt << 4) + mrow) << 5) + (half << 3);
      v4u lo = *(const v4u*)bp, hi = *(const v4u*)(bp + 16);
      v8u bu = {lo[0], lo[1], lo[2], lo[3], hi[0], hi[1], hi[2], hi[3]};
      return __builtin_bit_cast(v16bf, bu);
    };
    a0 = __builtin_amdgcn_wmma_f32_16x16x32_bf16(false, av, false, loadB(0), (short)0, a0, false, false);
    a1 = __builtin_amdgcn_wmma_f32_16x16x32_bf16(false, av, false, loadB(1), (short)0, a1, false, false);
    a2 = __builtin_amdgcn_wmma_f32_16x16x32_bf16(false, av, false, loadB(2), (short)0, a2, false, false);
    a3 = __builtin_amdgcn_wmma_f32_16x16x32_bf16(false, av, false, loadB(3), (short)0, a3, false, false);
  }

  int ncol = lane & 15;
#pragma unroll
  for (int r = 0; r < 8; ++r) {
    int mm = row0 + half * 8 + r;
    long long rb = cOff + (long long)mm * ldc + col0 + ncol;
#define EPILOG(ACC, NT)                                                        \
    {                                                                          \
      float vv = ACC[r] * alpha;                                               \
      if (bias) vv += bias[col0 + ncol + (NT) * 16];                           \
      if (relu) vv = fmaxf(vv, 0.0f);                                          \
      if (Cf)  Cf [rb + (NT) * 16] = vv;                                       \
      if (Cbf) Cbf[rb + (NT) * 16] = f2bf(vv);                                 \
    }
    EPILOG(a0, 0) EPILOG(a1, 1) EPILOG(a2, 2) EPILOG(a3, 3)
#undef EPILOG
  }
}

// ---------------------------------------------------------------------------
// Direct-global variant (for tiny M, e.g. fusion head M=16). One wave = 16x64.
// ---------------------------------------------------------------------------
__global__ __launch_bounds__(256) void gemm_bf16_wmma(
    const unsigned short* __restrict__ A, long long lda,
    const unsigned short* __restrict__ Bt, long long ldb,
    const float* __restrict__ bias,
    float* __restrict__ Cf, unsigned short* __restrict__ Cbf, long long ldc,
    int M, int N, int K, float alpha, int relu, int zdiv,
    long long sA1, long long sA2, long long sB1, long long sB2,
    long long sC1, long long sC2)
{
  int z  = blockIdx.z;
  int zq = z / zdiv, zr = z % zdiv;
  A  += (long long)zq * sA1 + (long long)zr * sA2;
  Bt += (long long)zq * sB1 + (long long)zr * sB2;
  long long cOff = (long long)zq * sC1 + (long long)zr * sC2;

  int wid  = threadIdx.x >> 5;
  int lane = threadIdx.x & 31;
  int tilesM = M >> 4;
  int strips = tilesM * (N >> 6);
  int strip  = blockIdx.x * 8 + wid;
  if (strip >= strips) return;
  int tm = strip % tilesM, tn = strip / tilesM;
  int row0 = tm << 4, col0 = tn << 6;
  int mrow = lane & 15, half = lane >> 4;

  v8f a0 = {}, a1 = {}, a2 = {}, a3 = {};
  const unsigned short* aBase = A  + (long long)(row0 + mrow) * lda + half * 8;
  const unsigned short* bBase = Bt + (long long)(col0 + mrow) * ldb + half * 8;

  for (int k0 = 0; k0 < K; k0 += 32) {
    const unsigned short* ap = aBase + k0;
    v8u au;
    au[0] = *(const unsigned int*)(ap + 0);
    au[1] = *(const unsigned int*)(ap + 2);
    au[2] = *(const unsigned int*)(ap + 4);
    au[3] = *(const unsigned int*)(ap + 6);
    au[4] = *(const unsigned int*)(ap + 16);
    au[5] = *(const unsigned int*)(ap + 18);
    au[6] = *(const unsigned int*)(ap + 20);
    au[7] = *(const unsigned int*)(ap + 22);
    v16bf av = __builtin_bit_cast(v16bf, au);

    auto loadB = [&](int nt) -> v16bf {
      const unsigned short* bp = bBase + (long long)nt * 16 * ldb + k0;
      v8u bu;
#pragma unroll
      for (int j = 0; j < 8; ++j) bu[j] = *(const unsigned int*)(bp + 2 * j + (j >= 4 ? 8 : 0));
      return __builtin_bit_cast(v16bf, bu);
    };
    a0 = __builtin_amdgcn_wmma_f32_16x16x32_bf16(false, av, false, loadB(0), (short)0, a0, false, false);
    a1 = __builtin_amdgcn_wmma_f32_16x16x32_bf16(false, av, false, loadB(1), (short)0, a1, false, false);
    a2 = __builtin_amdgcn_wmma_f32_16x16x32_bf16(false, av, false, loadB(2), (short)0, a2, false, false);
    a3 = __builtin_amdgcn_wmma_f32_16x16x32_bf16(false, av, false, loadB(3), (short)0, a3, false, false);
  }

  int ncol = lane & 15;
#pragma unroll
  for (int r = 0; r < 8; ++r) {
    int mm = row0 + half * 8 + r;
    long long rb = cOff + (long long)mm * ldc + col0 + ncol;
#define EPILOG(ACC, NT)                                                        \
    {                                                                          \
      float vv = ACC[r] * alpha;                                               \
      if (bias) vv += bias[col0 + ncol + (NT) * 16];                           \
      if (relu) vv = fmaxf(vv, 0.0f);                                          \
      if (Cf)  Cf [rb + (NT) * 16] = vv;                                       \
      if (Cbf) Cbf[rb + (NT) * 16] = f2bf(vv);                                 \
    }
    EPILOG(a0, 0) EPILOG(a1, 1) EPILOG(a2, 2) EPILOG(a3, 3)
#undef EPILOG
  }
}

// ---------------------------------------------------------------------------
// Elementwise / reduction helpers (block = 256 threads)
// ---------------------------------------------------------------------------
__device__ __forceinline__ float blkSum(float v, float* sh) {
  int tid = threadIdx.x;
  sh[tid] = v; __syncthreads();
  for (int s = 128; s > 0; s >>= 1) { if (tid < s) sh[tid] += sh[tid + s]; __syncthreads(); }
  float r = sh[0]; __syncthreads();
  return r;
}
__device__ __forceinline__ float blkMax(float v, float* sh) {
  int tid = threadIdx.x;
  sh[tid] = v; __syncthreads();
  for (int s = 128; s > 0; s >>= 1) { if (tid < s) sh[tid] = fmaxf(sh[tid], sh[tid + s]); __syncthreads(); }
  float r = sh[0]; __syncthreads();
  return r;
}
__device__ __forceinline__ void ln512(float& v0, float& v1, float* sh) {
  float mean = blkSum(v0 + v1, sh) * (1.0f / 512.0f);
  float d0 = v0 - mean, d1 = v1 - mean;
  float var = blkSum(d0 * d0 + d1 * d1, sh) * (1.0f / 512.0f);
  float rstd = rsqrtf(var + 1e-5f);
  v0 = d0 * rstd; v1 = d1 * rstd;
}

__global__ __launch_bounds__(256) void zero_f32(float* p, long long n) {
  long long i = (long long)blockIdx.x * 256 + threadIdx.x;
  long long stride = (long long)gridDim.x * 256;
  for (; i < n; i += stride) p[i] = 0.0f;
}
__global__ __launch_bounds__(256) void conv_bf(const float* s, unsigned short* d, long long n) {
  long long i = (long long)blockIdx.x * 256 + threadIdx.x;
  long long stride = (long long)gridDim.x * 256;
  for (; i < n; i += stride) d[i] = f2bf(s[i]);
}
__global__ __launch_bounds__(256) void convT_bf(const float* src, unsigned short* dst,
                                                int K, int N, long long sStride, long long dStride) {
  src += (long long)blockIdx.y * sStride;
  dst += (long long)blockIdx.y * dStride;
  long long idx = (long long)blockIdx.x * 256 + threadIdx.x;
  if (idx >= (long long)K * N) return;
  int n = (int)(idx / K), k = (int)(idx % K);
  dst[idx] = f2bf(src[(long long)k * N + n]);
}
__global__ __launch_bounds__(256) void vtrans_bf(const float* v32, unsigned short* vt) {
  long long idx = (long long)blockIdx.x * 256 + threadIdx.x;
  long long tot = (long long)CB * CNH * CHD * CS;
  if (idx >= tot) return;
  int j = (int)(idx % CS); long long r = idx / CS;
  int d = (int)(r % CHD); r /= CHD;
  int h = (int)(r % CNH); int b = (int)(r / CNH);
  vt[idx] = f2bf(v32[(((long long)b * CS + j) * CNH + h) * CHD + d]);
}

__global__ __launch_bounds__(256) void embed_ln(
    const int* et, const int* cat, const int* nm, const int* qy,
    const float* price, const float* ts,
    const float* Ee, const float* Ec, const float* En, const float* Eq,
    const float* wp, const float* wt, float* feat32, unsigned short* featbf)
{
  __shared__ float sh[256];
  int tok = blockIdx.x, tid = threadIdx.x;
  int e = et[tok], c = cat[tok], n = nm[tok], q = qy[tok];
  float p = price[tok], t = ts[tok];
  int h0 = tid, h1 = tid + 256;
  float v0 = Ee[(long long)e * CH + h0] + Ec[(long long)c * CH + h0]
           + En[(long long)n * CH + h0] + Eq[(long long)q * CH + h0]
           + p * wp[h0] + t * wt[h0];
  float v1 = Ee[(long long)e * CH + h1] + Ec[(long long)c * CH + h1]
           + En[(long long)n * CH + h1] + Eq[(long long)q * CH + h1]
           + p * wp[h1] + t * wt[h1];
  ln512(v0, v1, sh);
  long long base = (long long)tok * CH;
  feat32[base + h0] = v0; feat32[base + h1] = v1;
  featbf[base + h0] = f2bf(v0); featbf[base + h1] = f2bf(v1);
}

__global__ __launch_bounds__(256) void enc_select(const float* gout, const int* et, int t,
                                                  float* feat2, unsigned short* feat2bf) {
  __shared__ float sh[256];
  int tok = blockIdx.x;
  if (et[tok] != t) return;
  int tid = threadIdx.x;
  long long base = (long long)tok * CH;
  float v0 = gout[base + tid], v1 = gout[base + tid + 256];
  ln512(v0, v1, sh);
  v0 = fmaxf(v0, 0.0f); v1 = fmaxf(v1, 0.0f);
  feat2[base + tid] = v0; feat2[base + tid + 256] = v1;
  feat2bf[base + tid] = f2bf(v0); feat2bf[base + tid + 256] = f2bf(v1);
}

__global__ __launch_bounds__(256) void add_ln(const float* a, const float* b,
                                              float* outF, unsigned short* outBF) {
  __shared__ float sh[256];
  int tok = blockIdx.x, tid = threadIdx.x;
  long long base = (long long)tok * CH;
  float v0 = a[base + tid] + b[base + tid];
  float v1 = a[base + tid + 256] + b[base + tid + 256];
  ln512(v0, v1, sh);
  outF[base + tid] = v0; outF[base + tid + 256] = v1;
  if (outBF) { outBF[base + tid] = f2bf(v0); outBF[base + tid + 256] = f2bf(v1); }
}

__global__ __launch_bounds__(256) void softmax_row(unsigned short* sc, const int* et,
                                                   const unsigned char* mask, int t) {
  __shared__ float sh[256];
  int row = blockIdx.x, tid = threadIdx.x;
  int z = row / CS;
  int b = z / CNH;
  unsigned short* p = sc + (long long)row * CS;
  int j0 = tid, j1 = tid + 256;
  bool kv0 = (et[(long long)b * CS + j0] == t) && (mask[(long long)b * CS + j0] != 0);
  bool kv1 = (et[(long long)b * CS + j1] == t) && (mask[(long long)b * CS + j1] != 0);
  float x0 = kv0 ? bf2f(p[j0]) : -1e9f;
  float x1 = kv1 ? bf2f(p[j1]) : -1e9f;
  float mx = blkMax(fmaxf(x0, x1), sh);
  float e0 = __expf(x0 - mx), e1 = __expf(x1 - mx);
  float sum = blkSum(e0 + e1, sh);
  float inv = 1.0f / fmaxf(sum, 1e-20f);
  p[j0] = f2bf(e0 * inv); p[j1] = f2bf(e1 * inv);
}

__global__ __launch_bounds__(256) void pool_scatter(const float* x2, const int* et,
                                                    const unsigned char* mask,
                                                    float* pooled_t, float* out, int t) {
  __shared__ int cposA[CS];
  __shared__ unsigned char kvs[CS], vls[CS];
  __shared__ int scnt;
  int b = blockIdx.x, tid = threadIdx.x;
  if (tid == 0) {
    int c = 0, kc = 0;
    for (int s = 0; s < CS; ++s) {
      bool valid = (et[(long long)b * CS + s] == t);
      bool kv = valid && (mask[(long long)b * CS + s] != 0);
      cposA[s] = c; vls[s] = valid ? 1 : 0; kvs[s] = kv ? 1 : 0;
      if (valid) ++c;
      if (kv) ++kc;
    }
    scnt = kc;
  }
  __syncthreads();
  float s0 = 0.0f, s1 = 0.0f;
  for (int s = 0; s < CS; ++s) {
    if (kvs[s]) {
      long long base = ((long long)b * CS + s) * CH;
      s0 += x2[base + tid]; s1 += x2[base + tid + 256];
    }
  }
  float inv = 1.0f / fmaxf((float)scnt, 1.0f);
  pooled_t[(long long)b * CH + tid] = s0 * inv;
  pooled_t[(long long)b * CH + tid + 256] = s1 * inv;
  for (int s = 0; s < CS; ++s) {
    if (vls[s]) {
      long long src = ((long long)b * CS + s) * CH;
      long long dst = ((long long)b * (CS + 1) + 1 + cposA[s]) * CH;
      out[dst + tid]       += x2[src + tid];
      out[dst + tid + 256] += x2[src + tid + 256];
    }
  }
}

__global__ __launch_bounds__(256) void concat_user(const float* pooled, unsigned short* userbf) {
  int idx = blockIdx.x * 256 + threadIdx.x;
  if (idx >= CB * CNT * CH) return;
  int b = idx / (CNT * CH);
  int c = idx % (CNT * CH);
  int t = c / CH, h = c % CH;
  userbf[idx] = f2bf(pooled[((long long)t * CB + b) * CH + h]);
}

__global__ __launch_bounds__(256) void ln_act(const float* in, float* outF, unsigned short* outBF,
                                              int L, int act, long long ldF, long long ldBF) {
  __shared__ float sh[256];
  int row = blockIdx.x, tid = threadIdx.x;
  int per = L >> 8;
  float v[4];
  const float* ip = in + (long long)row * L;
  float acc = 0.0f;
  for (int i = 0; i < per; ++i) { v[i] = ip[tid + i * 256]; acc += v[i]; }
  float mean = blkSum(acc, sh) / (float)L;
  acc = 0.0f;
  for (int i = 0; i < per; ++i) { v[i] -= mean; acc += v[i] * v[i]; }
  float rstd = rsqrtf(blkSum(acc, sh) / (float)L + 1e-5f);
  for (int i = 0; i < per; ++i) {
    float x = v[i] * rstd;
    if (act == 1) x = fmaxf(x, 0.0f);
    else          x = tanhf(x);
    int col = tid + i * 256;
    if (outF)  outF [(long long)row * ldF  + col] = x;
    if (outBF) outBF[(long long)row * ldBF + col] = f2bf(x);
  }
}

// ---------------------------------------------------------------------------
extern "C" void kernel_launch(void* const* d_in, const int* in_sizes, int n_in,
                              void* d_out, int out_size, void* d_ws, size_t ws_size,
                              hipStream_t stream) {
  (void)in_sizes; (void)n_in; (void)out_size; (void)ws_size;
  const int* et  = (const int*)d_in[0];
  const int* cat = (const int*)d_in[1];
  const int* nm  = (const int*)d_in[2];
  const int* qy  = (const int*)d_in[3];
  const unsigned char* mask = (const unsigned char*)d_in[4];
  const float* prices  = (const float*)d_in[5];
  const float* tstamps = (const float*)d_in[6];
  const float* Ee = (const float*)d_in[7];
  const float* Ec = (const float*)d_in[8];
  const float* En = (const float*)d_in[9];
  const float* Eq = (const float*)d_in[10];
  const float* wp = (const float*)d_in[11];
  const float* wt = (const float*)d_in[12];
  const float* W_enc = (const float*)d_in[13];
  const float* b_enc = (const float*)d_in[14];
  const float* Wq = (const float*)d_in[15];
  const float* Wk = (const float*)d_in[16];
  const float* Wv = (const float*)d_in[17];
  const float* Wo = (const float*)d_in[18];
  const float* bq = (const float*)d_in[19];
  const float* bk = (const float*)d_in[20];
  const float* bv = (const float*)d_in[21];
  const float* bo = (const float*)d_in[22];
  const float* W1 = (const float*)d_in[23];
  const float* b1 = (const float*)d_in[24];
  const float* W2 = (const float*)d_in[25];
  const float* b2 = (const float*)d_in[26];
  const float* Wf1 = (const float*)d_in[27];
  const float* bf1 = (const float*)d_in[28];
  const float* Wf2 = (const float*)d_in[29];
  const float* bf2 = (const float*)d_in[30];
  float* out = (float*)d_out;

  char* base = (char*)d_ws;
  size_t off = 0;
  auto alloc = [&](size_t bytes) -> void* {
    off = (off + 255) & ~(size_t)255;
    void* p = base + off; off += bytes; return p;
  };
  typedef unsigned short u16;
  u16* wencb = (u16*)alloc((size_t)CNT * CH * CH * 2);
  u16* wqb   = (u16*)alloc((size_t)CNT * CH * CH * 2);
  u16* wkb   = (u16*)alloc((size_t)CNT * CH * CH * 2);
  u16* wvb   = (u16*)alloc((size_t)CNT * CH * CH * 2);
  u16* wob   = (u16*)alloc((size_t)CNT * CH * CH * 2);
  u16* w1t   = (u16*)alloc((size_t)CNT * CH * CFF * 2);
  u16* w2t   = (u16*)alloc((size_t)CNT * CFF * CH * 2);
  u16* wf1t  = (u16*)alloc((size_t)(CNT * CH) * (2 * CH) * 2);
  u16* wf2t  = (u16*)alloc((size_t)(2 * CH) * CH * 2);
  float* feat32  = (float*)alloc((size_t)CTOK * CH * 4);
  u16*   featbf  = (u16*)  alloc((size_t)CTOK * CH * 2);
  float* gout32  = (float*)alloc((size_t)CTOK * CH * 4);
  float* feat2_32= (float*)alloc((size_t)CTOK * CH * 4);
  u16*   feat2bf = (u16*)  alloc((size_t)CTOK * CH * 2);
  float* q32     = (float*)alloc((size_t)CTOK * CH * 4);
  float* k32     = (float*)alloc((size_t)CTOK * CH * 4);
  float* v32     = (float*)alloc((size_t)CTOK * CH * 4);
  u16*   qbf     = (u16*)  alloc((size_t)CTOK * CH * 2);
  u16*   kbf     = (u16*)  alloc((size_t)CTOK * CH * 2);
  u16*   vtbf    = (u16*)  alloc((size_t)CTOK * CH * 2);
  u16*   scoresbf= (u16*)  alloc((size_t)CB * CNH * CS * CS * 2);
  float* attno32 = (float*)alloc((size_t)CTOK * CH * 4);
  u16*   attnobf = (u16*)  alloc((size_t)CTOK * CH * 2);
  float* x1_32   = (float*)alloc((size_t)CTOK * CH * 4);
  u16*   x1bf    = (u16*)  alloc((size_t)CTOK * CH * 2);
  u16*   ffn1bf  = (u16*)  alloc((size_t)CTOK * CFF * 2);
  float* x2_32   = (float*)alloc((size_t)CTOK * CH * 4);
  float* pooled  = (float*)alloc((size_t)CNT * CB * CH * 4);
  u16*   userbf  = (u16*)  alloc((size_t)CB * CNT * CH * 2);
  u16*   h1bf    = (u16*)  alloc((size_t)CB * 2 * CH * 2);

  // LDS-staged GEMM (M % 128 == 0)
  auto gemmL = [&](const u16* A, long long lda, const u16* Bt, long long ldb,
                   const float* bias, float* Cf, u16* Cbf, long long ldc,
                   int M, int N, int K, float alpha, int relu,
                   int batches, int zdiv,
                   long long sA1, long long sA2, long long sB1, long long sB2,
                   long long sC1, long long sC2) {
    dim3 grid((M / 128) * (N / 64), 1, batches);
    gemm_bf16_lds<<<grid, 256, 0, stream>>>(A, lda, Bt, ldb, bias, Cf, Cbf, ldc,
                                            M, N, K, alpha, relu, zdiv,
                                            sA1, sA2, sB1, sB2, sC1, sC2);
  };
  // direct-global GEMM (any M % 16 == 0)
  auto gemmS = [&](const u16* A, long long lda, const u16* Bt, long long ldb,
                   const float* bias, float* Cf, u16* Cbf, long long ldc,
                   int M, int N, int K, float alpha, int relu) {
    int strips = (M / 16) * (N / 64);
    dim3 grid((strips + 7) / 8, 1, 1);
    gemm_bf16_wmma<<<grid, 256, 0, stream>>>(A, lda, Bt, ldb, bias, Cf, Cbf, ldc,
                                             M, N, K, alpha, relu, 1, 0, 0, 0, 0, 0, 0);
  };

  // 0) zero output (tf accumulates into rows 1..S)
  zero_f32<<<256, 256, 0, stream>>>(out, (long long)CB * (CS + 1) * CH);

  // 1) weight convert + transpose to [N][K] bf16
  convT_bf<<<dim3((CH * CH + 255) / 256, CNT), 256, 0, stream>>>(W_enc, wencb, CH, CH, (long long)CH * CH, (long long)CH * CH);
  convT_bf<<<dim3((CH * CH + 255) / 256, CNT), 256, 0, stream>>>(Wq, wqb, CH, CH, (long long)CH * CH, (long long)CH * CH);
  convT_bf<<<dim3((CH * CH + 255) / 256, CNT), 256, 0, stream>>>(Wk, wkb, CH, CH, (long long)CH * CH, (long long)CH * CH);
  convT_bf<<<dim3((CH * CH + 255) / 256, CNT), 256, 0, stream>>>(Wv, wvb, CH, CH, (long long)CH * CH, (long long)CH * CH);
  convT_bf<<<dim3((CH * CH + 255) / 256, CNT), 256, 0, stream>>>(Wo, wob, CH, CH, (long long)CH * CH, (long long)CH * CH);
  convT_bf<<<dim3((CH * CFF + 255) / 256, CNT), 256, 0, stream>>>(W1, w1t, CH, CFF, (long long)CH * CFF, (long long)CH * CFF);
  convT_bf<<<dim3((CFF * CH + 255) / 256, CNT), 256, 0, stream>>>(W2, w2t, CFF, CH, (long long)CFF * CH, (long long)CFF * CH);
  convT_bf<<<dim3(((CNT * CH) * (2 * CH) + 255) / 256, 1), 256, 0, stream>>>(Wf1, wf1t, CNT * CH, 2 * CH, 0, 0);
  convT_bf<<<dim3(((2 * CH) * CH + 255) / 256, 1), 256, 0, stream>>>(Wf2, wf2t, 2 * CH, CH, 0, 0);

  // 2) embedding + LN
  embed_ln<<<CTOK, 256, 0, stream>>>(et, cat, nm, qy, prices, tstamps,
                                     Ee, Ec, En, Eq, wp, wt, feat32, featbf);

  // 3) per-type encoders + masked select
  for (int t = 0; t < CNT; ++t) {
    gemmL(featbf, CH, wencb + (size_t)t * CH * CH, CH, b_enc + (size_t)t * CH,
          gout32, nullptr, CH, CTOK, CH, CH, 1.0f, 0, 1, 1, 0, 0, 0, 0, 0, 0);
    enc_select<<<CTOK, 256, 0, stream>>>(gout32, et, t, feat2_32, feat2bf);
  }

  // 4) per-type transformer block
  for (int t = 0; t < CNT; ++t) {
    gemmL(feat2bf, CH, wqb + (size_t)t * CH * CH, CH, bq + (size_t)t * CH,
          q32, nullptr, CH, CTOK, CH, CH, 1.0f, 0, 1, 1, 0, 0, 0, 0, 0, 0);
    gemmL(feat2bf, CH, wkb + (size_t)t * CH * CH, CH, bk + (size_t)t * CH,
          k32, nullptr, CH, CTOK, CH, CH, 1.0f, 0, 1, 1, 0, 0, 0, 0, 0, 0);
    gemmL(feat2bf, CH, wvb + (size_t)t * CH * CH, CH, bv + (size_t)t * CH,
          v32, nullptr, CH, CTOK, CH, CH, 1.0f, 0, 1, 1, 0, 0, 0, 0, 0, 0);
    conv_bf<<<1024, 256, 0, stream>>>(q32, qbf, (long long)CTOK * CH);
    conv_bf<<<1024, 256, 0, stream>>>(k32, kbf, (long long)CTOK * CH);
    vtrans_bf<<<(CTOK * CH + 255) / 256, 256, 0, stream>>>(v32, vtbf);

    // scores = (Q K^T)/8 per (b,h), bf16 out
    gemmL(qbf, CH, kbf, CH, nullptr, nullptr, scoresbf, CS,
          CS, CS, CHD, 0.125f, 0, CB * CNH, CNH,
          (long long)CS * CH, CHD, (long long)CS * CH, CHD,
          (long long)CNH * CS * CS, (long long)CS * CS);
    softmax_row<<<CB * CNH * CS, 256, 0, stream>>>(scoresbf, et, mask, t);
    // O = P V per (b,h)
    gemmL(scoresbf, CS, vtbf, CS, nullptr, attno32, nullptr, CH,
          CS, CHD, CS, 1.0f, 0, CB * CNH, CNH,
          (long long)CNH * CS * CS, (long long)CS * CS,
          (long long)CNH * CHD * CS, (long long)CHD * CS,
          (long long)CS * CH, CHD);
    conv_bf<<<1024, 256, 0, stream>>>(attno32, attnobf, (long long)CTOK * CH);

    gemmL(attnobf, CH, wob + (size_t)t * CH * CH, CH, bo + (size_t)t * CH,
          gout32, nullptr, CH, CTOK, CH, CH, 1.0f, 0, 1, 1, 0, 0, 0, 0, 0, 0);
    add_ln<<<CTOK, 256, 0, stream>>>(feat2_32, gout32, x1_32, x1bf);

    gemmL(x1bf, CH, w1t + (size_t)t * CH * CFF, CH, b1 + (size_t)t * CFF,
          nullptr, ffn1bf, CFF, CTOK, CFF, CH, 1.0f, 1, 1, 1, 0, 0, 0, 0, 0, 0);
    gemmL(ffn1bf, CFF, w2t + (size_t)t * CFF * CH, CFF, b2 + (size_t)t * CH,
          gout32, nullptr, CH, CTOK, CH, CFF, 1.0f, 0, 1, 1, 0, 0, 0, 0, 0, 0);
    add_ln<<<CTOK, 256, 0, stream>>>(x1_32, gout32, x2_32, nullptr);

    pool_scatter<<<CB, 256, 0, stream>>>(x2_32, et, mask,
                                         pooled + (size_t)t * CB * CH, out, t);
  }

  // 5) fusion head (M=16 -> direct-global GEMM)
  concat_user<<<(CB * CNT * CH + 255) / 256, 256, 0, stream>>>(pooled, userbf);
  gemmS(userbf, CNT * CH, wf1t, CNT * CH, bf1, gout32, nullptr, 2 * CH,
        CB, 2 * CH, CNT * CH, 1.0f, 0);
  ln_act<<<CB, 256, 0, stream>>>(gout32, nullptr, h1bf, 2 * CH, 1, 2 * CH, 2 * CH);
  gemmS(h1bf, 2 * CH, wf2t, 2 * CH, bf2, gout32, nullptr, CH,
        CB, CH, 2 * CH, 1.0f, 0);
  ln_act<<<CB, 256, 0, stream>>>(gout32, out, nullptr, CH, 2, (long long)(CS + 1) * CH, CH);
}
